// SelfAttention_40888088658185
// MI455X (gfx1250) — matrix operand order repaired and checked by reading, not compiled, not run on hardware
//
#include <hip/hip_runtime.h>
#include <hip/hip_bf16.h>

// ---------------------------------------------------------------------------
// CDNA5 (gfx1250) self-attention: bf16 WMMA everywhere, fp32 accumulate.
// B=8, C=512, H=W=32 -> seq=1024, heads=8, head_size=64.
// GEMMs use double-buffered GLOBAL_LOAD_ASYNC_TO_LDS_B128 staging (ASYNCcnt)
// when the toolchain exposes the builtins; otherwise register staging.
// ---------------------------------------------------------------------------

typedef unsigned short u16;
typedef __attribute__((ext_vector_type(4)))  unsigned int u32x4;   // 16B vector
typedef __attribute__((ext_vector_type(4)))  int          v4i;     // 16B vector
typedef __attribute__((ext_vector_type(16))) __bf16       v16bf;
typedef __attribute__((ext_vector_type(8)))  float        v8f;

#if defined(__gfx1250__) &&                                            \
    __has_builtin(__builtin_amdgcn_global_load_async_to_lds_b128) &&   \
    __has_builtin(__builtin_amdgcn_s_wait_asynccnt)
#define HAVE_ASYNC 1
#else
#define HAVE_ASYNC 0
#endif

#define AS1 __attribute__((address_space(1)))
#define AS3 __attribute__((address_space(3)))

union Frag {            // one WMMA A/B operand: 16 bf16 per lane = 32 bytes
  v16bf v;
  u32x4 q[2];
};

__device__ __forceinline__ u16 f2bf(float f) {
  union { float f; unsigned u; } cv; cv.f = f;
  unsigned u = cv.u;
  u += 0x7FFFu + ((u >> 16) & 1u);     // round-to-nearest-even
  return (u16)(u >> 16);
}

__device__ __forceinline__ v8f wmma_bf16(const Frag& a, const Frag& b, v8f c) {
  // D = A*B + C   (16x16x32, bf16 in, f32 accumulate)
  return __builtin_amdgcn_wmma_f32_16x16x32_bf16(
      false, a.v, false, b.v, (short)0, c, false, false);
}

// 16B global -> LDS copy, per lane. Async (no VGPR data path) when available.
__device__ __forceinline__ void gload_lds_b128(const u16* g, u16* l) {
#if HAVE_ASYNC
  __builtin_amdgcn_global_load_async_to_lds_b128(
      (AS1 v4i*)g, (AS3 v4i*)l, /*offset=*/0, /*cpol=*/0);
#else
  *(u32x4*)l = *(const u32x4*)g;
#endif
}
__device__ __forceinline__ void wait_async_all() {
#if HAVE_ASYNC
  __builtin_amdgcn_s_wait_asynccnt(0);
#endif
}

// ---- problem constants ----
#define BATCH 8
#define CH    512
#define SEQ   1024
#define NH    8
#define HS    64
#define MTOT  (BATCH * SEQ)        // 8192

// ---------------------------------------------------------------------------
// Prep: x (B,C,S) f32 -> xT (B,S,C) bf16
// ---------------------------------------------------------------------------
__global__ void k_prep_x(const float* __restrict__ x, u16* __restrict__ xT) {
  int idx = blockIdx.x * 256 + threadIdx.x;          // B*S*C = 4,194,304
  if (idx >= BATCH * SEQ * CH) return;
  int c = idx & (CH - 1);
  int s = (idx >> 9) & (SEQ - 1);
  int b = idx >> 19;
  xT[idx] = f2bf(x[(b * CH + c) * SEQ + s]);
}

// W (K,N) f32 -> WT (N,K) bf16
__global__ void k_transpose_w(const float* __restrict__ W, u16* __restrict__ WT,
                              int K, int N) {
  int idx = blockIdx.x * 256 + threadIdx.x;
  if (idx >= K * N) return;
  int k = idx % K, n = idx / K;
  WT[idx] = f2bf(W[k * N + n]);
}

// ---------------------------------------------------------------------------
// Shared 128x128 (K=512) bf16 WMMA main loop, BK=32, 8 waves (2 x 4).
// Double-buffered async global->LDS staging; copy of tile k+1 overlaps the
// 8 WMMAs of tile k; s_wait_asynccnt 0 + barrier at the iteration boundary.
// ---------------------------------------------------------------------------
__device__ __forceinline__ void gemm_tile_128x128(
    const u16* __restrict__ Ag,       // (M, 512) bf16, row-major
    const u16* __restrict__ Bg,       // (N, 512) bf16 (= B^T), row-major
    int m0, int n0, v8f acc[4][2],
    u16 (&ldsA)[2][128][40], u16 (&ldsB)[2][128][40]) {
  const int tid  = threadIdx.x;
  const int lane = tid & 31, wv = tid >> 5;
  const int wm = wv >> 2, wn = wv & 3;
  const int r = lane & 15, hi = lane >> 4;
  const int lrow = tid >> 1, lcol = (tid & 1) * 16;

  const u16* gA = &Ag[(m0 + lrow) * 512 + lcol];
  const u16* gB = &Bg[(n0 + lrow) * 512 + lcol];

  // prologue: stage K-tile 0 into buffer 0
  gload_lds_b128(gA,     &ldsA[0][lrow][lcol]);
  gload_lds_b128(gA + 8, &ldsA[0][lrow][lcol + 8]);
  gload_lds_b128(gB,     &ldsB[0][lrow][lcol]);
  gload_lds_b128(gB + 8, &ldsB[0][lrow][lcol + 8]);
  wait_async_all();
  __syncthreads();

  for (int kt = 0; kt < 16; ++kt) {
    const int cur = kt & 1, nxt = cur ^ 1;
    if (kt + 1 < 16) {                     // issue next tile's copy (overlap)
      const u16* pa = gA + (kt + 1) * 32;
      const u16* pb = gB + (kt + 1) * 32;
      gload_lds_b128(pa,     &ldsA[nxt][lrow][lcol]);
      gload_lds_b128(pa + 8, &ldsA[nxt][lrow][lcol + 8]);
      gload_lds_b128(pb,     &ldsB[nxt][lrow][lcol]);
      gload_lds_b128(pb + 8, &ldsB[nxt][lrow][lcol + 8]);
      if (kt + 2 < 16)                     // cache hint two tiles ahead
        __builtin_prefetch(gA + (kt + 2) * 32, 0, 1);
    }

    Frag afr[4], bfr[2];
#pragma unroll
    for (int mf = 0; mf < 4; ++mf) {
      int row = wm * 64 + mf * 16 + r;
      afr[mf].q[0] = *(const u32x4*)&ldsA[cur][row][hi * 8];
      afr[mf].q[1] = *(const u32x4*)&ldsA[cur][row][16 + hi * 8];
    }
#pragma unroll
    for (int nf = 0; nf < 2; ++nf) {
      int row = wn * 32 + nf * 16 + r;
      bfr[nf].q[0] = *(const u32x4*)&ldsB[cur][row][hi * 8];
      bfr[nf].q[1] = *(const u32x4*)&ldsB[cur][row][16 + hi * 8];
    }
#pragma unroll
    for (int mf = 0; mf < 4; ++mf)
#pragma unroll
      for (int nf = 0; nf < 2; ++nf)
        acc[mf][nf] = wmma_bf16(afr[mf], bfr[nf], acc[mf][nf]);

    wait_async_all();      // next tile resident in LDS
    __syncthreads();       // all waves done reading cur / writes visible
  }
}

// ---------------------------------------------------------------------------
// QKV GEMM: (8192 x 512) @ (512 x 1536) + bias.
//   epilogue scatters: Q,K row-major (bh, s, d) bf16 ; V transposed (bh, d, s)
// ---------------------------------------------------------------------------
__global__ __launch_bounds__(256) void k_qkv_gemm(
    const u16* __restrict__ xT,     // (8192, 512)
    const u16* __restrict__ WT,     // (1536, 512)  (= Wqkv^T)
    const float* __restrict__ bqkv, // (1536)
    u16* __restrict__ Qb, u16* __restrict__ Kb, u16* __restrict__ Vt) {
  __shared__ __align__(16) u16 ldsA[2][128][40];
  __shared__ __align__(16) u16 ldsB[2][128][40];

  const int lane = threadIdx.x & 31, wv = threadIdx.x >> 5;
  const int wm = wv >> 2, wn = wv & 3;
  const int r = lane & 15, hi = lane >> 4;
  const int m0 = blockIdx.x * 128;
  const int n0 = blockIdx.y * 128;

  v8f zero = {};
  v8f acc[4][2];
#pragma unroll
  for (int i = 0; i < 4; ++i)
#pragma unroll
    for (int j = 0; j < 2; ++j) acc[i][j] = zero;

  gemm_tile_128x128(xT, WT, m0, n0, acc, ldsA, ldsB);

  // epilogue: D element e -> (m = e + 8*hi, n = lane&15)
#pragma unroll
  for (int mf = 0; mf < 4; ++mf)
#pragma unroll
    for (int nf = 0; nf < 2; ++nf)
#pragma unroll
      for (int e = 0; e < 8; ++e) {
        int m = m0 + wm * 64 + mf * 16 + e + 8 * hi;
        int n = n0 + wn * 32 + nf * 16 + r;
        float val = acc[mf][nf][e] + bqkv[n];
        int b = m >> 10, s = m & (SEQ - 1);
        int part = n >> 9, cc = n & (CH - 1);
        int h = cc >> 6, d = cc & 63;
        int bh = b * NH + h;
        u16 bv = f2bf(val);
        if (part == 0)      Qb[(bh * SEQ + s) * HS + d] = bv;
        else if (part == 1) Kb[(bh * SEQ + s) * HS + d] = bv;
        else                Vt[(bh * HS + d) * SEQ + s] = bv;   // transposed
      }
}

// ---------------------------------------------------------------------------
// Flash attention: block = (bh, qtile of 64); 4 waves x 16 query rows.
// Per key tile of 32: S = Q K^T (4 WMMAs), online softmax (f32), O += P V
// (4 WMMAs). P is re-laid-out C->A via wave-private LDS staging.
// ---------------------------------------------------------------------------
__global__ __launch_bounds__(128) void k_attn(
    const u16* __restrict__ Qb, const u16* __restrict__ Kb,
    const u16* __restrict__ Vt, u16* __restrict__ O /* (b, s, 512) bf16 */) {
  __shared__ __align__(16) u16 ldsP[4][16][32];

  const int tid = threadIdx.x, lane = tid & 31, wv = tid >> 5;
  const int r = lane & 15, hi = lane >> 4;
  const int bh = blockIdx.x;
  const int q0 = blockIdx.y * 64 + wv * 16;

  const u16* Qrow = Qb + (size_t)bh * SEQ * HS;
  const u16* Krow = Kb + (size_t)bh * SEQ * HS;
  const u16* Vrow = Vt + (size_t)bh * HS * SEQ;

  // Q A-fragments (d0 = 0, 32), rows = this wave's 16 queries
  Frag qa[2];
  {
    const u16* p = Qrow + (q0 + r) * HS;
#pragma unroll
    for (int dh = 0; dh < 2; ++dh) {
      qa[dh].q[0] = *(const u32x4*)(p + dh * 32 + hi * 8);
      qa[dh].q[1] = *(const u32x4*)(p + dh * 32 + 16 + hi * 8);
    }
  }

  v8f zero = {};
  v8f oacc[4];
#pragma unroll
  for (int f = 0; f < 4; ++f) oacc[f] = zero;
  float mrun[8], lrun[8];
#pragma unroll
  for (int e = 0; e < 8; ++e) { mrun[e] = -1e30f; lrun[e] = 0.0f; }

  for (int kt = 0; kt < SEQ / 32; ++kt) {
    const int kbase = kt * 32;
    // K B-fragments: [key half nh][d half dh]
    Frag kfr[2][2];
#pragma unroll
    for (int nh = 0; nh < 2; ++nh) {
      const u16* p = Krow + (kbase + nh * 16 + r) * HS;
#pragma unroll
      for (int dh = 0; dh < 2; ++dh) {
        kfr[nh][dh].q[0] = *(const u32x4*)(p + dh * 32 + hi * 8);
        kfr[nh][dh].q[1] = *(const u32x4*)(p + dh * 32 + 16 + hi * 8);
      }
    }
    v8f s0 = zero, s1 = zero;
    s0 = wmma_bf16(qa[0], kfr[0][0], s0);
    s0 = wmma_bf16(qa[1], kfr[0][1], s0);
    s1 = wmma_bf16(qa[0], kfr[1][0], s1);
    s1 = wmma_bf16(qa[1], kfr[1][1], s1);

    // online softmax per local row (row = e + 8*hi, replicated over 16 lanes)
#pragma unroll
    for (int e = 0; e < 8; ++e) {
      float a  = s0[e] * 0.125f;          // 1/sqrt(64)
      float b2 = s1[e] * 0.125f;
      float tmax = fmaxf(a, b2);
#pragma unroll
      for (int off = 1; off < 16; off <<= 1)
        tmax = fmaxf(tmax, __shfl_xor(tmax, off, 32));
      float nm = fmaxf(mrun[e], tmax);
      float sf = __expf(mrun[e] - nm);
      float p0 = __expf(a - nm);
      float p1 = __expf(b2 - nm);
      float rs = p0 + p1;
#pragma unroll
      for (int off = 1; off < 16; off <<= 1)
        rs += __shfl_xor(rs, off, 32);
      lrun[e] = lrun[e] * sf + rs;
      mrun[e] = nm;
#pragma unroll
      for (int f = 0; f < 4; ++f) oacc[f][e] = oacc[f][e] * sf;
      // stage P (C-layout -> LDS row-major 16x32)
      ldsP[wv][e + 8 * hi][r]      = f2bf(p0);
      ldsP[wv][e + 8 * hi][16 + r] = f2bf(p1);
    }
    // re-load P as A-fragment (same wave; backend inserts s_wait_dscnt)
    Frag pa;
    pa.q[0] = *(const u32x4*)&ldsP[wv][r][hi * 8];
    pa.q[1] = *(const u32x4*)&ldsP[wv][r][16 + hi * 8];

    // O += P @ V  (V stored transposed: rows = d, contiguous in s)
#pragma unroll
    for (int f = 0; f < 4; ++f) {
      Frag vb;
      const u16* p = Vrow + (f * 16 + r) * SEQ + kbase + hi * 8;
      vb.q[0] = *(const u32x4*)p;
      vb.q[1] = *(const u32x4*)(p + 16);
      oacc[f] = wmma_bf16(pa, vb, oacc[f]);
    }
  }

  const int b = bh >> 3, h = bh & 7;
#pragma unroll
  for (int f = 0; f < 4; ++f)
#pragma unroll
    for (int e = 0; e < 8; ++e) {
      int q = q0 + e + 8 * hi;
      int d = f * 16 + r;
      float val = oacc[f][e] / lrun[e];
      O[((size_t)b * SEQ + q) * CH + h * HS + d] = f2bf(val);
    }
}

// ---------------------------------------------------------------------------
// Output projection: (8192 x 512) @ (512 x 512) + bias, write f32 (b, c, s)
// ---------------------------------------------------------------------------
__global__ __launch_bounds__(256) void k_out_gemm(
    const u16* __restrict__ A,      // attn out (8192, 512) bf16
    const u16* __restrict__ WT,     // Wout^T (512, 512) bf16
    const float* __restrict__ bout, // (512)
    float* __restrict__ out) {      // (8, 512, 1024) f32
  __shared__ __align__(16) u16 ldsA[2][128][40];
  __shared__ __align__(16) u16 ldsB[2][128][40];

  const int lane = threadIdx.x & 31, wv = threadIdx.x >> 5;
  const int wm = wv >> 2, wn = wv & 3;
  const int r = lane & 15, hi = lane >> 4;
  const int m0 = blockIdx.x * 128;
  const int n0 = blockIdx.y * 128;

  v8f zero = {};
  v8f acc[4][2];
#pragma unroll
  for (int i = 0; i < 4; ++i)
#pragma unroll
    for (int j = 0; j < 2; ++j) acc[i][j] = zero;

  gemm_tile_128x128(A, WT, m0, n0, acc, ldsA, ldsB);

#pragma unroll
  for (int mf = 0; mf < 4; ++mf)
#pragma unroll
    for (int nf = 0; nf < 2; ++nf)
#pragma unroll
      for (int e = 0; e < 8; ++e) {
        int m = m0 + wm * 64 + mf * 16 + e + 8 * hi;
        int n = n0 + wn * 32 + nf * 16 + r;
        float val = acc[mf][nf][e] + bout[n];
        int b = m >> 10, s = m & (SEQ - 1);
        out[((size_t)b * CH + n) * SEQ + s] = val;   // (b, c, h, w)
      }
}

// ---------------------------------------------------------------------------
// Launch
// ---------------------------------------------------------------------------
extern "C" void kernel_launch(void* const* d_in, const int* in_sizes, int n_in,
                              void* d_out, int out_size, void* d_ws,
                              size_t ws_size, hipStream_t stream) {
  const float* x    = (const float*)d_in[0];
  const float* Wqkv = (const float*)d_in[1];
  const float* bqkv = (const float*)d_in[2];
  const float* Wout = (const float*)d_in[3];
  const float* bout = (const float*)d_in[4];
  float* out = (float*)d_out;

  char* ws = (char*)d_ws;
  u16* xT    = (u16*)(ws + 0);         //  8 MB  (8192, 512) bf16
  u16* WqkvT = (u16*)(ws + 8388608);   //  1.5MB (1536, 512)
  u16* WoutT = (u16*)(ws + 9961472);   //  0.5MB (512, 512)
  u16* Qb    = (u16*)(ws + 10485760);  //  8 MB  (64, 1024, 64)
  u16* Kb    = (u16*)(ws + 18874368);  //  8 MB  (64, 1024, 64)
  u16* Vt    = (u16*)(ws + 27262976);  //  8 MB  (64, 64, 1024) transposed
  u16* aO    = (u16*)(ws + 35651584);  //  8 MB  (8192, 512)

  k_prep_x<<<(BATCH * SEQ * CH + 255) / 256, 256, 0, stream>>>(x, xT);
  k_transpose_w<<<(512 * 1536 + 255) / 256, 256, 0, stream>>>(Wqkv, WqkvT, 512, 1536);
  k_transpose_w<<<(512 * 512 + 255) / 256, 256, 0, stream>>>(Wout, WoutT, 512, 512);

  k_qkv_gemm<<<dim3(MTOT / 128, 1536 / 128), 256, 0, stream>>>(
      xT, WqkvT, bqkv, Qb, Kb, Vt);

  k_attn<<<dim3(BATCH * NH, SEQ / 64), 128, 0, stream>>>(Qb, Kb, Vt, aO);

  k_out_gemm<<<dim3(MTOT / 128, 512 / 128), 256, 0, stream>>>(
      aO, WoutT, bout, out);
}
